// TopM_MHSA_44495861187238
// MI455X (gfx1250) — compile-verified, matrix operand order checked by reading, not run on hardware
//
#include <hip/hip_runtime.h>
#include <hip/hip_bf16.h>
#include <math.h>
#include <stdint.h>

// ---------------- problem constants ----------------
#define LAYERS 2
#define BATCH  4
#define SEQ    1024
#define CDIM   256
#define HEADS  8
#define DHEAD  32          // CDIM / HEADS
#define TOPM   64
#define DFF    1024
#define ROWS   (BATCH * SEQ)      // 4096 token rows
#define LNEPS  1e-5f

// WMMA vector types (CDNA5 wave32)
typedef __attribute__((ext_vector_type(16))) __bf16 v16bf;
typedef __attribute__((ext_vector_type(8)))  float  v8f;

union ABOp {            // one 16x32 bf16 WMMA operand per lane
    uint16_t u[16];
    uint4    q[2];      // two 16-byte chunks
    v16bf    v;
};

__device__ __forceinline__ uint16_t f2bf(float f) {
    uint32_t u = __float_as_uint(f);
    uint32_t r = u + 0x7FFFu + ((u >> 16) & 1u);   // round-to-nearest-even
    return (uint16_t)(r >> 16);
}

__device__ __forceinline__ v8f wmma_bf16(const ABOp& a, const ABOp& b, v8f c) {
    return __builtin_amdgcn_wmma_f32_16x16x32_bf16(
        false, a.v, false, b.v, (short)0, c, false, false);
}

// ---------------- elementwise f32 -> bf16 convert (weights) ----------------
__global__ void cvt_bf16_kernel(const float* __restrict__ in,
                                uint16_t* __restrict__ out, int n) {
    int i = blockIdx.x * blockDim.x + threadIdx.x;
    if (i < n) out[i] = f2bf(in[i]);
}

// ---------------- LayerNorm (one 256-thread block per row) ----------------
__global__ void ln_kernel(const float* __restrict__ x,
                          const float* __restrict__ g,
                          const float* __restrict__ b,
                          uint16_t* __restrict__ out) {
    __shared__ float s1[8], s2[8], stats[2];
    const int row  = blockIdx.x;
    const int tid  = threadIdx.x;
    const int wave = tid >> 5, lane = tid & 31;
    float v = x[(size_t)row * CDIM + tid];
    float s = v, ss = v * v;
    #pragma unroll
    for (int off = 16; off >= 1; off >>= 1) {
        s  += __shfl_xor(s,  off, 32);
        ss += __shfl_xor(ss, off, 32);
    }
    if (lane == 0) { s1[wave] = s; s2[wave] = ss; }
    __syncthreads();
    if (tid == 0) {
        float a = 0.f, c = 0.f;
        #pragma unroll
        for (int i = 0; i < 8; ++i) { a += s1[i]; c += s2[i]; }
        float mean = a * (1.0f / CDIM);
        float var  = c * (1.0f / CDIM) - mean * mean;
        stats[0] = mean;
        stats[1] = rsqrtf(var + LNEPS);
    }
    __syncthreads();
    float y = (v - stats[0]) * stats[1] * g[tid] + b[tid];
    out[(size_t)row * CDIM + tid] = f2bf(y);
}

// ---------------- generic bf16 WMMA GEMM: D = act(A@B + bias) (+resid) ----
// A: [M,K] bf16 row-major.  B: [K,Nc] bf16 row-major (staged in LDS).
// grid = (Nc/64, M/32), block = 256 (8 waves, each owns a 16x16 output tile)
template<bool GELU_, bool RES, bool OUTBF>
__global__ void gemm_bf16_kernel(const uint16_t* __restrict__ A,
                                 const uint16_t* __restrict__ Bw,
                                 const float* __restrict__ bias,
                                 const float* __restrict__ resid,
                                 float* __restrict__ outf,
                                 uint16_t* __restrict__ outb,
                                 int M, int K, int Nc) {
    __shared__ uint16_t Bs[32 * 72];     // 32 K-rows x 64 cols (+8 pad)
    const int tid  = threadIdx.x;
    const int wave = tid >> 5, lane = tid & 31;
    const int g    = lane >> 4;          // half-wave select
    const int wm   = wave >> 2, wn = wave & 3;
    const int row0 = blockIdx.y * 32 + wm * 16;
    const int col0 = blockIdx.x * 64;
    const int colT = col0 + wn * 16;
    const int rB   = tid >> 3, ccB = (tid & 7) * 8;   // B-stage mapping

    v8f acc = {};
    #pragma unroll 2
    for (int kk = 0; kk < K; kk += 32) {
        __syncthreads();
        // cooperative stage of B tile (32x64 bf16 = 4KB, coalesced b128)
        *(uint4*)(&Bs[rB * 72 + ccB]) =
            *(const uint4*)(&Bw[(size_t)(kk + rB) * Nc + col0 + ccB]);
        // prefetch next K-tile of B into cache (gfx1250 global_prefetch path)
        if (kk + 32 < K)
            __builtin_prefetch(&Bw[(size_t)(kk + 32 + rB) * Nc + col0 + ccB], 0, 1);
        __syncthreads();
        // A operand: row m = row0+lane%16, K-chunks at kk+g*8 and kk+16+g*8
        ABOp a;
        const uint16_t* ar = A + (size_t)(row0 + (lane & 15)) * K + kk;
        a.q[0] = *(const uint4*)(ar + g * 8);
        a.q[1] = *(const uint4*)(ar + 16 + g * 8);
        // B operand from LDS: col n = colT+lane%16, same K pattern
        ABOp bm;
        const int n = wn * 16 + (lane & 15);
        #pragma unroll
        for (int j = 0; j < 16; ++j) {
            int kloc = (j < 8) ? (g * 8 + j) : (16 + g * 8 + (j - 8));
            bm.u[j] = Bs[kloc * 72 + n];
        }
        acc = wmma_bf16(a, bm, acc);
    }
    // C/D layout: element v -> (m = row0 + g*8 + v, n = colT + lane%16)
    const int nc = colT + (lane & 15);
    const float bb = bias[nc];
    #pragma unroll
    for (int v = 0; v < 8; ++v) {
        int m = row0 + g * 8 + v;
        float r = acc[v] + bb;
        if (GELU_) r = 0.5f * r * (1.0f + erff(r * 0.70710678118654752f));
        if (RES)   r += resid[(size_t)m * Nc + nc];
        if (OUTBF) outb[(size_t)m * Nc + nc] = f2bf(r);
        else       outf[(size_t)m * Nc + nc] = r;
    }
}

// ---------------- fused dual-pass attention ----------------
// One workgroup = (b, h, 64-query panel). Scores stay in LDS (64 x 1032 f32),
// plus an 8KB LDS chunk for staged V (128 keys x 32 d, bf16).
// Phase 1: QK^T via WMMA. Phase 2: dense softmax stats + top-64 threshold by
// value bisection + fused combined probabilities. Phase 3: P@V via WMMA with
// LDS-staged V chunks (coalesced b128 staging instead of 2B gathers).
#define QBLK  64
#define SLD   1032    // 1024 + 8 pad floats to dodge bank conflicts
#define VCH   128     // keys per staged V chunk
__global__ void attn_kernel(const uint16_t* __restrict__ qb,   // [ROWS,CDIM] bf16
                            const uint16_t* __restrict__ kvb,  // [ROWS,2*CDIM] bf16
                            const float* __restrict__ wf,      // w_fuse[2]
                            uint16_t* __restrict__ attn) {     // [ROWS,CDIM] bf16
    extern __shared__ float sc[];
    uint16_t* Vs = (uint16_t*)(sc + QBLK * SLD);   // [VCH][DHEAD] bf16
    const int bid  = blockIdx.x;
    const int b    = bid / (HEADS * (SEQ / QBLK));
    const int rem  = bid % (HEADS * (SEQ / QBLK));
    const int h    = rem / (SEQ / QBLK);
    const int q0   = (rem % (SEQ / QBLK)) * QBLK;
    const int tid  = threadIdx.x;
    const int wave = tid >> 5, lane = tid & 31;
    const int g    = lane >> 4;
    const float scale = 0.17677669529663687f;   // 1/sqrt(32)

    // learned fusion weights (softmax over 2) -- uniform, scalarizes to SALU
    const float e0 = expf(wf[0]), e1 = expf(wf[1]);
    const float w0 = e0 / (e0 + e1), w1 = e1 / (e0 + e1);

    // ---- phase 1: scores tile 64q x 1024k, 256 WMMA tiles over 8 waves ----
    for (int t = wave; t < (QBLK / 16) * (SEQ / 16); t += 8) {
        const int qt = t >> 6, kt = t & 63;
        ABOp a, bm;
        {   // Q operand: 16 queries x 32 d (contiguous d chunks)
            int m = q0 + qt * 16 + (lane & 15);
            const uint16_t* qr = qb + (size_t)(b * SEQ + m) * CDIM + h * DHEAD;
            a.q[0] = *(const uint4*)(qr + g * 8);
            a.q[1] = *(const uint4*)(qr + 16 + g * 8);
        }
        {   // K operand: 16 keys across lanes, d packed (row-major contiguous)
            int n = kt * 16 + (lane & 15);
            const uint16_t* kr = kvb + (size_t)(b * SEQ + n) * (2 * CDIM) + h * DHEAD;
            bm.q[0] = *(const uint4*)(kr + g * 8);
            bm.q[1] = *(const uint4*)(kr + 16 + g * 8);
        }
        v8f c = {};
        c = wmma_bf16(a, bm, c);
        #pragma unroll
        for (int v = 0; v < 8; ++v)
            sc[(qt * 16 + g * 8 + v) * SLD + kt * 16 + (lane & 15)] = c[v] * scale;
    }
    __syncthreads();

    // ---- phase 2: per-row dense softmax + top-64 threshold + fusion ----
    for (int i = 0; i < 8; ++i) {
        const int row = wave * 8 + i;
        float* rp = sc + row * SLD;
        float rv[32];
        #pragma unroll
        for (int j = 0; j < 32; ++j) rv[j] = rp[lane + j * 32];
        float mx = -1e30f, mn = 1e30f;
        #pragma unroll
        for (int j = 0; j < 32; ++j) { mx = fmaxf(mx, rv[j]); mn = fminf(mn, rv[j]); }
        #pragma unroll
        for (int off = 16; off >= 1; off >>= 1) {
            mx = fmaxf(mx, __shfl_xor(mx, off, 32));
            mn = fminf(mn, __shfl_xor(mn, off, 32));
        }
        float sd = 0.f;
        #pragma unroll
        for (int j = 0; j < 32; ++j) sd += expf(rv[j] - mx);
        #pragma unroll
        for (int off = 16; off >= 1; off >>= 1) sd += __shfl_xor(sd, off, 32);
        // bisection for top-64 threshold: largest t with count(s>t) >= TOPM
        float lo = mn - 1.0f, hi = mx;
        for (int it = 0; it < 24; ++it) {
            float mid = 0.5f * (lo + hi);
            int c = 0;
            #pragma unroll
            for (int j = 0; j < 32; ++j) c += (rv[j] > mid) ? 1 : 0;
            #pragma unroll
            for (int off = 16; off >= 1; off >>= 1) c += __shfl_xor(c, off, 32);
            if (c >= TOPM) lo = mid; else hi = mid;
        }
        const float thr = lo;
        float st = 0.f;
        #pragma unroll
        for (int j = 0; j < 32; ++j) st += (rv[j] > thr) ? expf(rv[j] - mx) : 0.f;
        #pragma unroll
        for (int off = 16; off >= 1; off >>= 1) st += __shfl_xor(st, off, 32);
        const float cd = w0 / sd;
        const float ct = w1 / fmaxf(st, 1e-20f);
        #pragma unroll
        for (int j = 0; j < 32; ++j) {
            float e = expf(rv[j] - mx);
            rp[lane + j * 32] = e * (cd + ((rv[j] > thr) ? ct : 0.f));
        }
    }
    __syncthreads();

    // ---- phase 3: out = P @ V with LDS-staged V chunks ----
    {
        const int qt = wave >> 1, dt = wave & 1;
        const int d  = dt * 16 + (lane & 15);
        const int m  = qt * 16 + (lane & 15);
        v8f acc = {};
        for (int kk0 = 0; kk0 < SEQ; kk0 += VCH) {
            // stage V[kk0 .. kk0+VCH) x DHEAD : 8KB, coalesced b128 loads
            {
                const uint16_t* vb = kvb + (size_t)(b * SEQ + kk0) * (2 * CDIM)
                                     + CDIM + h * DHEAD;
                #pragma unroll
                for (int it = 0; it < 2; ++it) {
                    int idx = tid * 2 + it;            // 512 uint4 total
                    int r = idx >> 2, c = (idx & 3) * 8;
                    *(uint4*)(&Vs[r * DHEAD + c]) =
                        *(const uint4*)(vb + (size_t)r * (2 * CDIM) + c);
                }
            }
            __syncthreads();
            #pragma unroll
            for (int ks = 0; ks < VCH / 32; ++ks) {
                const int kk = kk0 + ks * 32;
                ABOp a, bm;
                #pragma unroll
                for (int j = 0; j < 16; ++j) {
                    int kl = (j < 8) ? (g * 8 + j) : (16 + g * 8 + (j - 8));
                    a.u[j]  = f2bf(sc[m * SLD + kk + kl]);
                    bm.u[j] = Vs[(ks * 32 + kl) * DHEAD + d];
                }
                acc = wmma_bf16(a, bm, acc);
            }
            __syncthreads();
        }
        #pragma unroll
        for (int v = 0; v < 8; ++v) {
            int mr = q0 + qt * 16 + g * 8 + v;
            int nc = dt * 16 + (lane & 15);
            attn[(size_t)(b * SEQ + mr) * CDIM + h * DHEAD + nc] = f2bf(acc[v]);
        }
    }
}

// ---------------- host side ----------------
extern "C" void kernel_launch(void* const* d_in, const int* in_sizes, int n_in,
                              void* d_out, int out_size, void* d_ws, size_t ws_size,
                              hipStream_t stream) {
    (void)in_sizes; (void)n_in; (void)out_size; (void)ws_size;
    const float* x     = (const float*)d_in[0];
    const float* ln1g  = (const float*)d_in[1];
    const float* ln1b  = (const float*)d_in[2];
    const float* Wq    = (const float*)d_in[3];
    const float* bq    = (const float*)d_in[4];
    const float* Wkv   = (const float*)d_in[5];
    const float* bkv   = (const float*)d_in[6];
    const float* wfuse = (const float*)d_in[7];
    const float* Wo    = (const float*)d_in[8];
    const float* bo    = (const float*)d_in[9];
    const float* ln2g  = (const float*)d_in[10];
    const float* ln2b  = (const float*)d_in[11];
    const float* W1    = (const float*)d_in[12];
    const float* b1    = (const float*)d_in[13];
    const float* W2    = (const float*)d_in[14];
    const float* b2    = (const float*)d_in[15];
    float* xcur = (float*)d_out;                       // running activations

    // workspace carve-up (bf16 buffers, 256B-aligned power-of-two sizes)
    uint8_t* ws = (uint8_t*)d_ws;
    uint16_t* wq_bf   = (uint16_t*)(ws);                 size_t off = CDIM * CDIM * 2;
    uint16_t* wkv_bf  = (uint16_t*)(ws + off);           off += CDIM * 2 * CDIM * 2;
    uint16_t* wo_bf   = (uint16_t*)(ws + off);           off += CDIM * CDIM * 2;
    uint16_t* w1_bf   = (uint16_t*)(ws + off);           off += CDIM * DFF * 2;
    uint16_t* w2_bf   = (uint16_t*)(ws + off);           off += DFF * CDIM * 2;
    uint16_t* h_bf    = (uint16_t*)(ws + off);           off += (size_t)ROWS * CDIM * 2;
    uint16_t* q_bf    = (uint16_t*)(ws + off);           off += (size_t)ROWS * CDIM * 2;
    uint16_t* kv_bf   = (uint16_t*)(ws + off);           off += (size_t)ROWS * 2 * CDIM * 2;
    uint16_t* attn_bf = (uint16_t*)(ws + off);           off += (size_t)ROWS * CDIM * 2;
    uint16_t* u_bf    = (uint16_t*)(ws + off);

    // x -> d_out (running buffer)
    hipMemcpyAsync(xcur, x, (size_t)ROWS * CDIM * sizeof(float),
                   hipMemcpyDeviceToDevice, stream);

    const int attnLds = QBLK * SLD * (int)sizeof(float)           // score panel
                      + VCH * DHEAD * (int)sizeof(uint16_t);      // V chunk
    hipFuncSetAttribute((const void*)attn_kernel,
                        hipFuncAttributeMaxDynamicSharedMemorySize, attnLds);

    const dim3 blk(256);
    for (int l = 0; l < LAYERS; ++l) {
        // per-layer weight slices -> bf16
        auto cvt = [&](const float* src, uint16_t* dst, int n) {
            cvt_bf16_kernel<<<dim3((n + 255) / 256), blk, 0, stream>>>(src, dst, n);
        };
        cvt(Wq  + (size_t)l * CDIM * CDIM,     wq_bf,  CDIM * CDIM);
        cvt(Wkv + (size_t)l * CDIM * 2 * CDIM, wkv_bf, CDIM * 2 * CDIM);
        cvt(Wo  + (size_t)l * CDIM * CDIM,     wo_bf,  CDIM * CDIM);
        cvt(W1  + (size_t)l * CDIM * DFF,      w1_bf,  CDIM * DFF);
        cvt(W2  + (size_t)l * DFF * CDIM,      w2_bf,  DFF * CDIM);

        // --- attention sublayer ---
        ln_kernel<<<dim3(ROWS), blk, 0, stream>>>(xcur, ln1g + l * CDIM,
                                                  ln1b + l * CDIM, h_bf);
        gemm_bf16_kernel<false, false, true>
            <<<dim3(CDIM / 64, ROWS / 32), blk, 0, stream>>>(
                h_bf, wq_bf, bq + l * CDIM, nullptr, nullptr, q_bf,
                ROWS, CDIM, CDIM);
        gemm_bf16_kernel<false, false, true>
            <<<dim3(2 * CDIM / 64, ROWS / 32), blk, 0, stream>>>(
                h_bf, wkv_bf, bkv + l * 2 * CDIM, nullptr, nullptr, kv_bf,
                ROWS, CDIM, 2 * CDIM);
        attn_kernel<<<dim3(BATCH * HEADS * (SEQ / QBLK)), blk, attnLds, stream>>>(
                q_bf, kv_bf, wfuse + l * 2, attn_bf);
        gemm_bf16_kernel<false, true, false>
            <<<dim3(CDIM / 64, ROWS / 32), blk, 0, stream>>>(
                attn_bf, wo_bf, bo + l * CDIM, xcur, xcur, nullptr,
                ROWS, CDIM, CDIM);

        // --- MLP sublayer ---
        ln_kernel<<<dim3(ROWS), blk, 0, stream>>>(xcur, ln2g + l * CDIM,
                                                  ln2b + l * CDIM, h_bf);
        gemm_bf16_kernel<true, false, true>
            <<<dim3(DFF / 64, ROWS / 32), blk, 0, stream>>>(
                h_bf, w1_bf, b1 + l * DFF, nullptr, nullptr, u_bf,
                ROWS, CDIM, DFF);
        gemm_bf16_kernel<false, true, false>
            <<<dim3(CDIM / 64, ROWS / 32), blk, 0, stream>>>(
                u_bf, w2_bf, b2 + l * CDIM, xcur, xcur, nullptr,
                ROWS, DFF, CDIM);
    }
}
